// RPN_4157528342622
// MI455X (gfx1250) — compile-verified, matrix operand order checked by reading, not run on hardware
//
#include <hip/hip_runtime.h>

// ---------- problem constants ----------
#define NB      8
#define CIN     512
#define CMID    512
#define HH      64
#define WW      64
#define NPIX    4096            // 64*64
#define NANCH   9
#define KTOT    36864           // NPIX*NANCH
#define NPRE    12000
#define NPOST   600
#define CANDMAX 12288
#define NMS_IOU 0.7f
#define MINSZ   16.0f

// K ordering for the implicit GEMM: k = kpos*512 + ci  (kpos = ky*3+kx)
#define KCHUNKS 144             // 4608/32
#define PPLANE  4356            // 66*66 padded plane (pixels)

// ---------- d_out layout (floats) ----------
#define OFF_LOC   0
#define OFF_SCORE (NB*KTOT*4)                 // 1179648
#define OFF_ROIS  (OFF_SCORE + NB*KTOT*2)     // 1769472
#define OFF_RIDX  (OFF_ROIS + NB*NPOST*4)     // 1788672
#define OFF_ANC   (OFF_RIDX + NB*NPOST)       // 1793472

// ---------- workspace layout (bytes) ----------
// xpack: [n][cblk=64][py*66+px][4 dwords] -> 8*64*4356*16 B
#define WS_XPACK 0u
#define SZ_XPACK (8u*64u*4356u*16u)                    // 35,684,352
#define WS_WPACK (WS_XPACK + SZ_XPACK)
#define SZ_WPACK (32u*144u*32u*8u*4u)                  // 4,718,592
#define WS_F     (WS_WPACK + SZ_WPACK)
#define SZ_F     (8u*512u*4096u*4u)                    // 67,108,864
#define WS_FG    (WS_F + SZ_F)
#define SZ_FG    (8u*36864u*4u)
#define WS_BOX   (WS_FG + SZ_FG)
#define SZ_BOX   (8u*36864u*4u*4u)
#define WS_SC    (WS_BOX + SZ_BOX)
#define SZ_SC    (8u*36864u*4u)
#define WS_CAND  (WS_SC + SZ_SC)

typedef __attribute__((ext_vector_type(16))) __bf16 v16bf;
typedef __attribute__((ext_vector_type(8)))  float  v8f;

union FragU { unsigned int u[8]; uint4 q[2]; v16bf v; };

__device__ __forceinline__ unsigned int f2bf(float f) {
    unsigned int u = __float_as_uint(f);
    unsigned int r = u + 0x7FFFu + ((u >> 16) & 1u);   // round-to-nearest-even
    return r >> 16;
}
__device__ __forceinline__ unsigned int packbf(float a, float b) {
    return (f2bf(a) & 0xFFFFu) | (f2bf(b) << 16);
}

// ---------------------------------------------------------------------------
// 1) pack x -> bf16, 8-channel groups contiguous per pixel, 66x66 zero halo.
//    xpack[((n*64+cblk)*4356 + py*66+px)] = uint4 of channels cblk*8..cblk*8+7
//    => a 32-deep-K B fragment is exactly two b128 loads per lane.
// ---------------------------------------------------------------------------
__global__ __launch_bounds__(256) void pack_x_kernel(const float* __restrict__ x,
                                                     uint4* __restrict__ xp) {
    unsigned int id = blockIdx.x * 256u + threadIdx.x;      // 2,230,272 total
    unsigned int pid = id % PPLANE;
    unsigned int pl  = id / PPLANE;                         // n*64 + cblk
    unsigned int cb  = pl & 63u;
    unsigned int n   = pl >> 6;
    int py = (int)(pid / 66u), px = (int)(pid % 66u);
    int y = py - 1, xx = px - 1;
    uint4 v = {0u, 0u, 0u, 0u};
    if (y >= 0 && y < HH && xx >= 0 && xx < WW) {
        size_t b = ((size_t)(n * CIN + cb * 8u)) * NPIX + (size_t)y * WW + xx;
        v.x = packbf(x[b],             x[b + 1u * NPIX]);
        v.y = packbf(x[b + 2u * NPIX], x[b + 3u * NPIX]);
        v.z = packbf(x[b + 4u * NPIX], x[b + 5u * NPIX]);
        v.w = packbf(x[b + 6u * NPIX], x[b + 7u * NPIX]);
    }
    xp[id] = v;
}

// ---------------------------------------------------------------------------
// 2) pack conv1_w into WMMA A-fragment order:
//    wpack[((mtile*144 + kc)*32 + lane)*8 + v] = pair (k0,k0+1) where
//    k0 = (v&3)*2 + (lane>=16)*8 + (v>>2)*16  within the 32-deep chunk.
// ---------------------------------------------------------------------------
__global__ __launch_bounds__(256) void pack_w_kernel(const float* __restrict__ w,
                                                     unsigned int* __restrict__ wp) {
    unsigned int id = blockIdx.x * 256u + threadIdx.x;      // 1,179,648 total
    unsigned int v    = id & 7u;
    unsigned int lane = (id >> 3) & 31u;
    unsigned int kc   = (id >> 8) % 144u;
    unsigned int mt   = id / (256u * 144u);                 // 0..31
    unsigned int g    = lane >> 4;
    unsigned int kl0  = (v & 3u) * 2u + g * 8u + (v >> 2) * 16u;
    unsigned int kpos = kc >> 4, cic = kc & 15u;
    unsigned int ci0  = cic * 32u + kl0;
    unsigned int co   = mt * 16u + (lane & 15u);
    size_t b0 = ((size_t)co * CIN + ci0) * 9u + kpos;
    size_t b1 = ((size_t)co * CIN + ci0 + 1u) * 9u + kpos;
    wp[id] = packbf(w[b0], w[b1]);
}

// ---------------------------------------------------------------------------
// 3) 3x3 conv as 9 shifted GEMMs, bf16 WMMA, fp32 accumulate, fused bias+ReLU.
//    grid: n(8) x yPair(32) x co-block(8); block: 256 thr = 8 waves.
//    wave -> 32(co) x 32(px): 2 A-frags x 2 B-frags -> 4 WMMA per K-chunk,
//    all operand loads are b128.
// ---------------------------------------------------------------------------
__global__ __launch_bounds__(256) void conv3_wmma_kernel(const uint4* __restrict__ Xp,
                                                         const uint4* __restrict__ Wp,
                                                         const float* __restrict__ bias,
                                                         float* __restrict__ F) {
    int blk  = blockIdx.x;
    int mb   = blk & 7;                 // co block of 64
    int y2   = (blk >> 3) & 31;         // row pair
    int n    = blk >> 8;
    int lane = threadIdx.x & 31;
    int wave = threadIdx.x >> 5;
    int mg   = wave & 1;                // co 32-group within block
    int ng   = wave >> 1;               // 0..3 : (row, xhalf)
    int yrow = y2 * 2 + (ng >> 1);
    int g    = lane >> 4;
    int col  = lane & 15;
    int x0   = (ng & 1) * 32 + col;     // subtile 0 pixel x
    int x1   = x0 + 16;                 // subtile 1 pixel x
    int mtile0 = mb * 4 + mg * 2;       // co tiles of 16
    int mtile1 = mtile0 + 1;

    v8f acc00 = {}; v8f acc01 = {}; v8f acc10 = {}; v8f acc11 = {};

    // A base in uint4 units: mtile*144*32*8/4 = mtile*9216 ; +kc*64 ; +lane*2
    const size_t a0base = (size_t)mtile0 * 9216u + (size_t)lane * 2u;
    const size_t a1base = (size_t)mtile1 * 9216u + (size_t)lane * 2u;
    const size_t xnbase = (size_t)n * 64u * PPLANE;     // uint4 units

    for (int kc = 0; kc < KCHUNKS; ++kc) {
        int kpos = kc >> 4, cic = kc & 15;
        int ky = kpos / 3, kx = kpos - 3 * ky;

        FragU a0, a1;
        const uint4* a0p = Wp + a0base + (size_t)kc * 64u;
        const uint4* a1p = Wp + a1base + (size_t)kc * 64u;
        a0.q[0] = a0p[0]; a0.q[1] = a0p[1];
        a1.q[0] = a1p[0]; a1.q[1] = a1p[1];
        if (kc + 1 < KCHUNKS) {
            __builtin_prefetch(Wp + a0base + (size_t)(kc + 1) * 64u, 0, 3);
            __builtin_prefetch(Wp + a1base + (size_t)(kc + 1) * 64u, 0, 3);
        }

        int rowoff = (yrow + ky) * 66 + kx;
        int cbl = cic * 4 + g;                          // low 8-channel block
        size_t p0 = xnbase + (size_t)cbl * PPLANE + rowoff;
        size_t p1 = p0 + 2u * PPLANE;                   // +16 channels

        FragU b0, b1;
        b0.q[0] = Xp[p0 + x0]; b0.q[1] = Xp[p1 + x0];
        b1.q[0] = Xp[p0 + x1]; b1.q[1] = Xp[p1 + x1];

        acc00 = __builtin_amdgcn_wmma_f32_16x16x32_bf16(false, a0.v, false, b0.v,
                                                        (short)0, acc00, false, false);
        acc01 = __builtin_amdgcn_wmma_f32_16x16x32_bf16(false, a0.v, false, b1.v,
                                                        (short)0, acc01, false, false);
        acc10 = __builtin_amdgcn_wmma_f32_16x16x32_bf16(false, a1.v, false, b0.v,
                                                        (short)0, acc10, false, false);
        acc11 = __builtin_amdgcn_wmma_f32_16x16x32_bf16(false, a1.v, false, b1.v,
                                                        (short)0, acc11, false, false);
    }

    // C layout: VGPR r -> M = r + g*8, N = lane&15
    size_t pix0 = (size_t)yrow * WW + x0;
    int coA = mtile0 * 16 + g * 8;
    int coB = mtile1 * 16 + g * 8;
#pragma unroll
    for (int r = 0; r < 8; ++r) {
        int c0 = coA + r, c1 = coB + r;
        float bv0 = bias[c0], bv1 = bias[c1];
        float o00 = acc00[r] + bv0; o00 = o00 > 0.0f ? o00 : 0.0f;
        float o01 = acc01[r] + bv0; o01 = o01 > 0.0f ? o01 : 0.0f;
        float o10 = acc10[r] + bv1; o10 = o10 > 0.0f ? o10 : 0.0f;
        float o11 = acc11[r] + bv1; o11 = o11 > 0.0f ? o11 : 0.0f;
        size_t f0 = ((size_t)(n * CMID + c0)) * NPIX;
        size_t f1 = ((size_t)(n * CMID + c1)) * NPIX;
        F[f0 + pix0]       = o00;
        F[f0 + pix0 + 16u] = o01;
        F[f1 + pix0]       = o10;
        F[f1 + pix0 + 16u] = o11;
    }
}

// ---------------------------------------------------------------------------
// 4) 1x1 heads (loc 36ch, score 18ch) + softmax fg, write rpn_locs/rpn_scores.
// ---------------------------------------------------------------------------
__global__ __launch_bounds__(256) void head1x1_kernel(const float* __restrict__ F,
                                                      const float* __restrict__ locw,
                                                      const float* __restrict__ locb,
                                                      const float* __restrict__ scw,
                                                      const float* __restrict__ scb,
                                                      float* __restrict__ out_loc,
                                                      float* __restrict__ out_sc,
                                                      float* __restrict__ fg) {
    int t = blockIdx.x * 256 + threadIdx.x;     // 32768 total
    int n = t >> 12, pix = t & 4095;
    float aL[36]; float aS[18];
#pragma unroll
    for (int i = 0; i < 36; ++i) aL[i] = 0.0f;
#pragma unroll
    for (int i = 0; i < 18; ++i) aS[i] = 0.0f;
    const float* fp = F + (size_t)n * CMID * NPIX + pix;
    for (int ci = 0; ci < CIN; ++ci) {
        float fv = fp[(size_t)ci * NPIX];
#pragma unroll
        for (int oc = 0; oc < 36; ++oc) aL[oc] = fmaf(fv, locw[oc * CIN + ci], aL[oc]);
#pragma unroll
        for (int oc = 0; oc < 18; ++oc) aS[oc] = fmaf(fv, scw[oc * CIN + ci], aS[oc]);
    }
    size_t kb = (size_t)n * KTOT + (size_t)pix * NANCH;
#pragma unroll
    for (int a = 0; a < NANCH; ++a) {
#pragma unroll
        for (int d = 0; d < 4; ++d)
            out_loc[(kb + a) * 4 + d] = aL[a * 4 + d] + locb[a * 4 + d];
        float s0 = aS[a * 2 + 0] + scb[a * 2 + 0];
        float s1 = aS[a * 2 + 1] + scb[a * 2 + 1];
        out_sc[(kb + a) * 2 + 0] = s0;
        out_sc[(kb + a) * 2 + 1] = s1;
        fg[kb + a] = 1.0f / (1.0f + __expf(s0 - s1));
    }
}

// ---------------------------------------------------------------------------
// 5) decode boxes, clip, min-size mask; also emit anchor output (n==0).
// ---------------------------------------------------------------------------
__global__ __launch_bounds__(256) void decode_kernel(const float* __restrict__ loc,
                                                     const float* __restrict__ fg,
                                                     const int* __restrict__ imgh,
                                                     const int* __restrict__ imgw,
                                                     float* __restrict__ boxes,
                                                     float* __restrict__ sc,
                                                     float* __restrict__ anc) {
    int t = blockIdx.x * 256 + threadIdx.x;     // 294912 total
    int n = t / KTOT, k = t - n * KTOT;
    int a = k % NANCH, pix = k / NANCH;
    int yy = pix >> 6, xx = pix & 63;
    int ir = a / 3, is_ = a - 3 * ir;
    const float ratios[3] = {0.5f, 1.0f, 2.0f};
    const float scales[3] = {8.0f, 16.0f, 32.0f};
    float r = ratios[ir], s = scales[is_];
    float bh = 16.0f * s * __fsqrt_rn(r);
    float bw = 16.0f * s * __fsqrt_rn(1.0f / r);
    float gx = (float)xx * 16.0f, gy = (float)yy * 16.0f;
    float ax1 = gx - 0.5f * bw, ay1 = gy - 0.5f * bh;
    float ax2 = gx + 0.5f * bw, ay2 = gy + 0.5f * bh;
    if (n == 0) {
        anc[(size_t)k * 4 + 0] = ax1; anc[(size_t)k * 4 + 1] = ay1;
        anc[(size_t)k * 4 + 2] = ax2; anc[(size_t)k * 4 + 3] = ay2;
    }
    float aw = ax2 - ax1, ah = ay2 - ay1;
    float acx = ax1 + 0.5f * aw, acy = ay1 + 0.5f * ah;
    const float* L = loc + (size_t)t * 4;
    float cx = L[0] * aw + acx, cy = L[1] * ah + acy;
    float nw = __expf(L[2]) * aw, nh2 = __expf(L[3]) * ah;
    float Wf = (float)imgw[0], Hf = (float)imgh[0];
    float x1 = fminf(fmaxf(cx - 0.5f * nw, 0.0f), Wf);
    float y1 = fminf(fmaxf(cy - 0.5f * nh2, 0.0f), Hf);
    float x2 = fminf(fmaxf(cx + 0.5f * nw, 0.0f), Wf);
    float y2 = fminf(fmaxf(cy + 0.5f * nh2, 0.0f), Hf);
    boxes[(size_t)t * 4 + 0] = x1; boxes[(size_t)t * 4 + 1] = y1;
    boxes[(size_t)t * 4 + 2] = x2; boxes[(size_t)t * 4 + 3] = y2;
    bool valid = (x2 - x1 >= MINSZ) && (y2 - y1 >= MINSZ);
    sc[t] = valid ? fg[t] : -__builtin_inff();
}

// ---------------------------------------------------------------------------
// 6) propose: threshold bisection (== top-12000 cut), compact, greedy NMS.
//    One workgroup (1024 thr) per image; scores LDS-resident.
// ---------------------------------------------------------------------------
__global__ __launch_bounds__(1024) void propose_kernel(const float* __restrict__ boxes,
                                                       const float* __restrict__ sc,
                                                       float* __restrict__ cand,
                                                       float* __restrict__ rois,
                                                       float* __restrict__ ridx) {
    __shared__ float s_sc[CANDMAX];
    __shared__ float s_val[1024];
    __shared__ int   s_idx[1024];
    __shared__ int   s_cnt;
    __shared__ float s_box[5];
    __shared__ float s_fb[5];
    const float NEGINF = -__builtin_inff();

    int n = blockIdx.x, tid = threadIdx.x;
    const float* bsc = sc + (size_t)n * KTOT;
    const float* bbx = boxes + (size_t)n * KTOT * 4;
    float* cbox  = cand + (size_t)n * CANDMAX * 5;
    float* carea = cbox + (size_t)CANDMAX * 4;

    // --- bisection for the N_PRE-th largest score ---
    float lo = 0.0f, hi = 1.0f;
    for (int it = 0; it < 30; ++it) {
        float t = 0.5f * (lo + hi);
        int c = 0;
        for (int k = tid; k < KTOT; k += 1024) c += (bsc[k] >= t) ? 1 : 0;
        s_idx[tid] = c; __syncthreads();
        for (int st = 512; st > 0; st >>= 1) {
            if (tid < st) s_idx[tid] += s_idx[tid + st];
            __syncthreads();
        }
        int tot = s_idx[0];
        __syncthreads();
        if (tot >= NPRE) lo = t; else hi = t;
    }
    float thr = lo;

    // --- compact candidates ---
    for (int j = tid; j < CANDMAX; j += 1024) s_sc[j] = NEGINF;
    if (tid == 0) s_cnt = 0;
    __syncthreads();
    for (int k = tid; k < KTOT; k += 1024) {
        float v = bsc[k];
        if (v >= thr) {
            int p = atomicAdd(&s_cnt, 1);
            if (p < CANDMAX) {
                float x1 = bbx[(size_t)k * 4 + 0], y1 = bbx[(size_t)k * 4 + 1];
                float x2 = bbx[(size_t)k * 4 + 2], y2 = bbx[(size_t)k * 4 + 3];
                cbox[(size_t)p * 4 + 0] = x1; cbox[(size_t)p * 4 + 1] = y1;
                cbox[(size_t)p * 4 + 2] = x2; cbox[(size_t)p * 4 + 3] = y2;
                carea[p] = (x2 - x1) * (y2 - y1);
                s_sc[p] = v;
            }
        }
    }
    __syncthreads();
    int m = s_cnt < CANDMAX ? s_cnt : CANDMAX;

    // --- greedy NMS, 600 picks ---
    for (int it = 0; it < NPOST; ++it) {
        float bv = NEGINF; int bi = 0;
        for (int j = tid; j < m; j += 1024) {
            float v = s_sc[j];
            if (v > bv) { bv = v; bi = j; }
        }
        s_val[tid] = bv; s_idx[tid] = bi;
        __syncthreads();
        for (int st = 512; st > 0; st >>= 1) {
            if (tid < st) {
                float ov = s_val[tid + st]; int oi = s_idx[tid + st];
                if (ov > s_val[tid] || (ov == s_val[tid] && oi < s_idx[tid])) {
                    s_val[tid] = ov; s_idx[tid] = oi;
                }
            }
            __syncthreads();
        }
        if (tid == 0) {
            float mv = s_val[0]; int mi = s_idx[0];
            if (mv > NEGINF) {
                s_box[0] = cbox[(size_t)mi * 4 + 0];
                s_box[1] = cbox[(size_t)mi * 4 + 1];
                s_box[2] = cbox[(size_t)mi * 4 + 2];
                s_box[3] = cbox[(size_t)mi * 4 + 3];
                s_box[4] = carea[mi];
            } else if (it == 0) {
                for (int q = 0; q < 5; ++q) s_box[q] = 0.0f;
            } else {
                for (int q = 0; q < 5; ++q) s_box[q] = s_fb[q];   // reference re-picks b[0]
            }
            if (it == 0)
                for (int q = 0; q < 5; ++q) s_fb[q] = s_box[q];
            size_t ro = ((size_t)n * NPOST + it) * 4;
            rois[ro + 0] = s_box[0]; rois[ro + 1] = s_box[1];
            rois[ro + 2] = s_box[2]; rois[ro + 3] = s_box[3];
        }
        __syncthreads();
        float X1 = s_box[0], Y1 = s_box[1], X2 = s_box[2], Y2 = s_box[3], AR = s_box[4];
        for (int j = tid; j < m; j += 1024) {
            float ix1 = fmaxf(X1, cbox[(size_t)j * 4 + 0]);
            float iy1 = fmaxf(Y1, cbox[(size_t)j * 4 + 1]);
            float ix2 = fminf(X2, cbox[(size_t)j * 4 + 2]);
            float iy2 = fminf(Y2, cbox[(size_t)j * 4 + 3]);
            float inter = fmaxf(ix2 - ix1, 0.0f) * fmaxf(iy2 - iy1, 0.0f);
            float iou = inter / fmaxf(AR + carea[j] - inter, 1e-12f);
            if (iou > NMS_IOU) s_sc[j] = NEGINF;
        }
        __syncthreads();
    }
    for (int it = tid; it < NPOST; it += 1024)
        ridx[(size_t)n * NPOST + it] = (float)n;
}

// ---------------------------------------------------------------------------
extern "C" void kernel_launch(void* const* d_in, const int* in_sizes, int n_in,
                              void* d_out, int out_size, void* d_ws, size_t ws_size,
                              hipStream_t stream) {
    const float* x       = (const float*)d_in[0];
    const float* conv1_w = (const float*)d_in[1];
    const float* conv1_b = (const float*)d_in[2];
    const float* score_w = (const float*)d_in[3];
    const float* score_b = (const float*)d_in[4];
    const float* loc_w   = (const float*)d_in[5];
    const float* loc_b   = (const float*)d_in[6];
    const int*   img_h   = (const int*)d_in[7];
    const int*   img_w   = (const int*)d_in[8];
    float* out = (float*)d_out;
    char*  ws  = (char*)d_ws;

    uint4*        xpack = (uint4*)(ws + WS_XPACK);
    unsigned int* wpack = (unsigned int*)(ws + WS_WPACK);
    float* F     = (float*)(ws + WS_F);
    float* fg    = (float*)(ws + WS_FG);
    float* boxes = (float*)(ws + WS_BOX);
    float* sc    = (float*)(ws + WS_SC);
    float* cand  = (float*)(ws + WS_CAND);

    pack_x_kernel<<<8712, 256, 0, stream>>>(x, xpack);
    pack_w_kernel<<<4608, 256, 0, stream>>>(conv1_w, wpack);
    conv3_wmma_kernel<<<2048, 256, 0, stream>>>(xpack, (const uint4*)wpack, conv1_b, F);
    head1x1_kernel<<<128, 256, 0, stream>>>(F, loc_w, loc_b, score_w, score_b,
                                            out + OFF_LOC, out + OFF_SCORE, fg);
    decode_kernel<<<1152, 256, 0, stream>>>(out + OFF_LOC, fg, img_h, img_w,
                                            boxes, sc, out + OFF_ANC);
    propose_kernel<<<8, 1024, 0, stream>>>(boxes, sc, cand,
                                           out + OFF_ROIS, out + OFF_RIDX);
}
